// _RPN_31988916420677
// MI455X (gfx1250) — matrix operand order repaired and checked by reading, not compiled, not run on hardware
//
#include <hip/hip_runtime.h>

// ---------------- problem constants ----------------
#define BATCH   2
#define CIN     512
#define COUT    512
#define TT      16
#define HH      32
#define WW      32
#define NPOS    16384            // T*H*W
#define KDIM    13824            // CIN*27
#define NIT     (KDIM / 32)      // 432 K-steps
#define NA      18
#define NANCH   294912           // NPOS*NA
#define TOPK    2000

// ---------------- workspace layout (bytes) ----------------
#define OFF_X     ((size_t)0)                       // bf16 activations: 32MB (k1..k2)
#define OFF_FB    ((size_t)33554432)                // bf16 feat: 32MB (k0..k1)
#define OFF_WB    ((size_t)67108864)                // bf16 weights: 13.5MB (k0..k1)
#define OFF_SC    ((size_t)33554432)                // scores f32 (aliases FB, k2+)
#define OFF_BX    ((size_t)35913728)                // boxes f32 (aliases FB, k2+)
#define OFF_HIST  ((size_t)81264640)                // 2*2048 int
#define OFF_META  ((size_t)81281024)                // 2*4 int
#define OFF_BCH   ((size_t)81281056)                // 2*288 int
#define OFF_BCE   ((size_t)81283360)                // 2*288 int

typedef __attribute__((ext_vector_type(16))) __bf16 v16bf;
typedef __attribute__((ext_vector_type(8)))  float  v8f;

union FragU { v16bf v; uint4 q[2]; };

__device__ __forceinline__ unsigned short f2bf_sw(float f) {
    unsigned u = __float_as_uint(f);
    unsigned r = u + 0x7FFFu + ((u >> 16) & 1u);   // round-to-nearest-even
    return (unsigned short)(r >> 16);
}
__device__ __forceinline__ unsigned pk2bf(float lo, float hi) {
#if __has_builtin(__builtin_amdgcn_cvt_pk_bf16_f32)
    typedef __attribute__((ext_vector_type(2))) __bf16 v2bf;
    union { v2bf v; unsigned u; } cv;
    cv.v = __builtin_amdgcn_cvt_pk_bf16_f32(lo, hi);
    return cv.u;
#else
    return (unsigned)f2bf_sw(lo) | ((unsigned)f2bf_sw(hi) << 16);
#endif
}
__device__ __forceinline__ unsigned short f2bf1(float f) {
#if __has_builtin(__builtin_amdgcn_cvt_pk_bf16_f32)
    return (unsigned short)(pk2bf(f, 0.f) & 0xFFFFu);
#else
    return f2bf_sw(f);
#endif
}
__device__ __forceinline__ unsigned sortkey(float f) {
    unsigned u = __float_as_uint(f);
    return (u & 0x80000000u) ? ~u : (u | 0x80000000u);
}
__device__ __forceinline__ float clampf(float v, float lo, float hi) {
    return fminf(fmaxf(v, lo), hi);
}

// async global->LDS copy of 32B per lane (CDNA5 Tensor/Async path, ASYNCcnt)
__device__ __forceinline__ void async_copy32(unsigned ldsAddr, const void* gptr) {
    asm volatile(
        "global_load_async_to_lds_b128 %0, %1, off\n\t"
        "global_load_async_to_lds_b128 %0, %1, off offset:16"
        :: "v"(ldsAddr), "v"(gptr) : "memory");
}
__device__ __forceinline__ void wait_async0() {
    asm volatile("s_wait_asynccnt 0x0" ::: "memory");
}

// anchor (w,h,dur): ratio-major {0.5,1,2} x scale {4,8,16}, each x dur {16,8}
__constant__ float ANCH_W[NA] = {92,92,184,184,368,368, 64,64,128,128,256,256, 44,44,88,88,176,176};
__constant__ float ANCH_H[NA] = {48,48,96,96,192,192, 64,64,128,128,256,256, 88,88,176,176,352,352};
__constant__ float ANCH_D[NA] = {16,8,16,8,16,8, 16,8,16,8,16,8, 16,8,16,8,16,8};

#define LDA_S 40   // LDS row stride (bf16 elems), 80B => 16B aligned
#define LDB_S 40

// =====================================================================
// Kernel 0: one-shot f32 -> bf16 conversion (weights AND feature map)
// =====================================================================
__global__ __launch_bounds__(256) void cvt_bf_kernel(
    const float* __restrict__ src, unsigned short* __restrict__ dst, int n)
{
    int base = (blockIdx.x * 256 + threadIdx.x) * 8;
    if (base < n) {
        float4 a = *(const float4*)(src + base);
        float4 b = *(const float4*)(src + base + 4);
        uint4 o;
        o.x = pk2bf(a.x, a.y); o.y = pk2bf(a.z, a.w);
        o.z = pk2bf(b.x, b.y); o.w = pk2bf(b.z, b.w);
        *(uint4*)(dst + base) = o;
    }
}

// =====================================================================
// Kernel 1: 3x3x3 conv (pad 1) as implicit GEMM, bf16 WMMA, fused bias+ReLU
//   Double-buffered: barrier -> issue next loads -> 8 WMMA -> store next.
//   A tile copied global->LDS via GLOBAL_LOAD_ASYNC_TO_LDS_B128 (no VGPR
//   round-trip, waits on ASYNCcnt after the WMMA block). B tile = masked
//   im2col gather with short-lived register staging inside the iteration.
// =====================================================================
__global__ __launch_bounds__(256, 1) void conv3d_wmma_kernel(
    const unsigned short* __restrict__ featb,
    const unsigned short* __restrict__ wconvb,
    const float* __restrict__ bconv, unsigned short* __restrict__ xout)
{
    __shared__ unsigned short As[2][128 * LDA_S];   // [buf][co][k]
    __shared__ unsigned short Bs[2][128 * LDB_S];   // [buf][pos][k]

    const int n0 = blockIdx.x * 128;
    const int m0 = blockIdx.y * 128;
    const int b  = blockIdx.z;
    const int tid  = threadIdx.x;
    const int wave = tid >> 5, lane = tid & 31;
    const int wm = wave >> 1, wn = wave & 1;
    const int lm = lane & 15, kh2 = lane >> 4;

    // A-staging mapping: (row, 16-elem k-half)
    const int rowLd  = tid >> 1;
    const int halfLd = tid & 1;
    // B-staging mapping: fixed k-lane, 16 consecutive positions
    const int klocal = tid >> 3;        // 0..31
    const int pg     = tid & 7;         // 0..7
    const int pbase  = n0 + pg * 16;
    const int pt = pbase >> 10, ph = (pbase >> 5) & 31, w0 = pbase & 31;

    const unsigned short* fb = featb + (size_t)b * CIN * NPOS;
    const unsigned short* wrow = wconvb + (size_t)(m0 + rowLd) * KDIM + halfLd * 16;
    const unsigned ldsA0 = (unsigned)(size_t)&As[0][rowLd * LDA_S + halfLd * 16];
    const unsigned ldsA1 = (unsigned)(size_t)&As[1][rowLd * LDA_S + halfLd * 16];

    v8f acc[2][4];
    #pragma unroll
    for (int i = 0; i < 2; ++i)
        #pragma unroll
        for (int j = 0; j < 4; ++j) {
            v8f z = {0.f,0.f,0.f,0.f,0.f,0.f,0.f,0.f};
            acc[i][j] = z;
        }

    // incremental tap state: k = klocal + 32*it ; ci = k/27, r = k%27
    int ci = klocal / 27;               // 0 or 1
    int r  = klocal - ci * 27;

    // ---- prologue: stage tile 0 into buffer 0 ----
    {
        async_copy32(ldsA0, wrow);
        int kt = r / 9, rr2 = r - kt * 9, kh = rr2 / 3, kw = rr2 - kh * 3;
        int st = pt + kt - 1, sh = ph + kh - 1;
        bool rowok = ((unsigned)st < TT) && ((unsigned)sh < HH);
        const unsigned short* rp = fb + (size_t)ci * NPOS + st * 1024 + sh * 32 + (w0 + kw - 1);
        #pragma unroll
        for (int i = 0; i < 16; ++i) {
            int sw = w0 + kw - 1 + i;
            Bs[0][(pg * 16 + i) * LDB_S + klocal] =
                (rowok && (unsigned)sw < WW) ? rp[i] : (unsigned short)0;
        }
        wait_async0();
    }

    for (int it = 0; it < NIT; ++it) {
        const int cur = it & 1;
        __syncthreads();                         // buf[cur] published to all waves

        // ---- issue NEXT tile's loads (A: async->LDS, B: ->regs) ----
        unsigned short bReg[16];
        const bool havenext = (it + 1 < NIT);
        if (havenext) {
            async_copy32(cur ? ldsA0 : ldsA1, wrow + (it + 1) * 32);
            r += 5; ci += 1;
            if (r >= 27) { r -= 27; ci += 1; }
            int kt = r / 9, rr2 = r - kt * 9, kh = rr2 / 3, kw = rr2 - kh * 3;
            int st = pt + kt - 1, sh = ph + kh - 1;
            bool rowok = ((unsigned)st < TT) && ((unsigned)sh < HH);
            const unsigned short* rp = fb + (size_t)ci * NPOS + st * 1024 + sh * 32 + (w0 + kw - 1);
            #pragma unroll
            for (int i = 0; i < 16; ++i) {
                int sw = w0 + kw - 1 + i;
                bReg[i] = (rowok && (unsigned)sw < WW) ? rp[i] : (unsigned short)0;
            }
        }

        // ---- fragments in contiguous k order (k = 16h + s) + 8 WMMA ----
        v16bf afr[2];
        #pragma unroll
        for (int i = 0; i < 2; ++i) {
            int row = wm * 32 + i * 16 + lm;
            FragU u;
            u.q[0] = *(const uint4*)&As[cur][row * LDA_S + kh2 * 16];      // K 0..7 / 16..23
            u.q[1] = *(const uint4*)&As[cur][row * LDA_S + kh2 * 16 + 8];  // K 8..15 / 24..31
            afr[i] = u.v;
        }
        #pragma unroll
        for (int j = 0; j < 4; ++j) {
            int col = wn * 64 + j * 16 + lm;
            FragU u;
            u.q[0] = *(const uint4*)&Bs[cur][col * LDB_S + kh2 * 16];      // K 0..15 / 16..31
            u.q[1] = *(const uint4*)&Bs[cur][col * LDB_S + kh2 * 16 + 8];
            acc[0][j] = __builtin_amdgcn_wmma_f32_16x16x32_bf16(
                false, afr[0], false, u.v, (short)0, acc[0][j], false, false);
            acc[1][j] = __builtin_amdgcn_wmma_f32_16x16x32_bf16(
                false, afr[1], false, u.v, (short)0, acc[1][j], false, false);
        }

        // ---- commit next B tile; wait for next A async copy ----
        if (havenext) {
            const int nxt = cur ^ 1;
            #pragma unroll
            for (int i = 0; i < 16; ++i)
                Bs[nxt][(pg * 16 + i) * LDB_S + klocal] = bReg[i];
            wait_async0();
        }
    }

    // ---- epilogue: bias + ReLU -> bf16 activations ----
    #pragma unroll
    for (int i = 0; i < 2; ++i) {
        int co0 = m0 + wm * 32 + i * 16 + kh2 * 8;
        #pragma unroll
        for (int j = 0; j < 4; ++j) {
            int pc = n0 + wn * 64 + j * 16 + lm;
            #pragma unroll
            for (int rr = 0; rr < 8; ++rr) {
                float v = acc[i][j][rr] + bconv[co0 + rr];
                v = v > 0.f ? v : 0.f;
                xout[((size_t)b * COUT + (co0 + rr)) * NPOS + pc] = f2bf1(v);
            }
        }
    }
}

// =====================================================================
// Kernel 2: fused cls/bbox 1x1x1 conv (bf16 WMMA) + softmax + box decode
// =====================================================================
__global__ __launch_bounds__(256, 1) void head_wmma_kernel(
    const unsigned short* __restrict__ xin,
    const float* __restrict__ wcls,  const float* __restrict__ bcls,
    const float* __restrict__ wbbox, const float* __restrict__ bbbox,
    const float* __restrict__ iminfo,
    float* __restrict__ scores, float* __restrict__ boxes)
{
    __shared__ __align__(32) unsigned char smem[87040];
    unsigned short* As2 = (unsigned short*)smem;            // 160*40 bf16 = 12800B
    unsigned short* Bs2 = (unsigned short*)(smem + 12800);  // 128*40 bf16 = 10240B
    float* Cs = (float*)smem;                               // aliased: 8*160*17*4 = 87040B

    const int n0 = blockIdx.x * 128;
    const int b  = blockIdx.y;
    const int tid  = threadIdx.x;
    const int wave = tid >> 5, lane = tid & 31;
    const int lm = lane & 15, kh2 = lane >> 4;

    v8f acc[10];
    #pragma unroll
    for (int m = 0; m < 10; ++m) { v8f z = {0.f,0.f,0.f,0.f,0.f,0.f,0.f,0.f}; acc[m] = z; }

    for (int k0 = 0; k0 < CIN; k0 += 32) {
        for (int idx = tid; idx < 320; idx += 256) {
            int row = idx >> 1, hf = idx & 1;
            const float* src = nullptr;
            if (row < 36)                      src = wcls  + (size_t)row * CIN + k0 + hf * 16;
            else if (row >= 48 && row < 156)   src = wbbox + (size_t)(row - 48) * CIN + k0 + hf * 16;
            unsigned* d = (unsigned*)&As2[row * LDA_S + hf * 16];
            #pragma unroll
            for (int i = 0; i < 8; ++i)
                d[i] = src ? pk2bf(src[2*i], src[2*i+1]) : 0u;
        }
        {
            int klocal = tid >> 3, pg = tid & 7;
            const unsigned short* src =
                xin + ((size_t)b * CIN + k0 + klocal) * NPOS + n0 + pg * 16;
            uint4 q0 = *(const uint4*)(src);
            uint4 q1 = *(const uint4*)(src + 8);
            unsigned v[8] = {q0.x,q0.y,q0.z,q0.w,q1.x,q1.y,q1.z,q1.w};
            #pragma unroll
            for (int i = 0; i < 8; ++i) {
                Bs2[(pg * 16 + 2*i    ) * LDB_S + klocal] = (unsigned short)(v[i] & 0xFFFFu);
                Bs2[(pg * 16 + 2*i + 1) * LDB_S + klocal] = (unsigned short)(v[i] >> 16);
            }
        }
        __syncthreads();

        v16bf bfr;
        {
            int col = wave * 16 + lm;
            FragU u;
            u.q[0] = *(const uint4*)&Bs2[col * LDB_S + kh2 * 16];
            u.q[1] = *(const uint4*)&Bs2[col * LDB_S + kh2 * 16 + 8];
            bfr = u.v;
        }
        #pragma unroll
        for (int mt = 0; mt < 10; ++mt) {
            int row = mt * 16 + lm;
            FragU u;
            u.q[0] = *(const uint4*)&As2[row * LDA_S + kh2 * 16];
            u.q[1] = *(const uint4*)&As2[row * LDA_S + kh2 * 16 + 8];
            acc[mt] = __builtin_amdgcn_wmma_f32_16x16x32_bf16(
                false, u.v, false, bfr, (short)0, acc[mt], false, false);
        }
        __syncthreads();
    }

    float* Cw = Cs + (size_t)wave * 160 * 17;
    #pragma unroll
    for (int mt = 0; mt < 10; ++mt) {
        #pragma unroll
        for (int rr = 0; rr < 8; ++rr) {
            int row = mt * 16 + kh2 * 8 + rr;
            float bias = 0.f;
            if (row < 36) bias = bcls[row];
            else if (row >= 48 && row < 156) bias = bbbox[row - 48];
            Cw[row * 17 + lm] = acc[mt][rr] + bias;
        }
    }
    __syncthreads();

    const int pos = lm;
    const int pp  = n0 + wave * 16 + pos;
    if (kh2 == 0) {
        #pragma unroll
        for (int a = 0; a < NA; ++a) {
            float l0 = Cw[a * 17 + pos];
            float l1 = Cw[(18 + a) * 17 + pos];
            float fg = 1.f / (1.f + __expf(l0 - l1));
            scores[(size_t)b * NANCH + (size_t)pp * NA + a] = fg;
        }
    } else {
        int t = pp >> 10, h = (pp >> 5) & 31, w = pp & 31;
        float cx = w * 16.f + 7.5f, cy = h * 16.f + 7.5f, ct = (float)t;
        float limx = iminfo[b * 3 + 1] - 1.f;
        float limy = iminfo[b * 3 + 0] - 1.f;
        float limt = (float)TT - 1.f;
        #pragma unroll
        for (int a = 0; a < NA; ++a) {
            const float aw = ANCH_W[a], ah = ANCH_H[a], ad = ANCH_D[a];
            int rb = 48 + a * 6;
            float d0 = Cw[(rb+0)*17 + pos], d1 = Cw[(rb+1)*17 + pos], d2 = Cw[(rb+2)*17 + pos];
            float d3 = Cw[(rb+3)*17 + pos], d4 = Cw[(rb+4)*17 + pos], d5 = Cw[(rb+5)*17 + pos];
            float pcx = d0 * aw + cx, pcy = d1 * ah + cy, pct = d2 * ad + ct;
            float psx = __expf(d3) * aw, psy = __expf(d4) * ah, pst = __expf(d5) * ad;
            float* o = boxes + ((size_t)b * NANCH + (size_t)pp * NA + a) * 6;
            o[0] = clampf(pcx - 0.5f * (psx - 1.f), 0.f, limx);
            o[1] = clampf(pcy - 0.5f * (psy - 1.f), 0.f, limy);
            o[2] = clampf(pct - 0.5f * (pst - 1.f), 0.f, limt);
            o[3] = clampf(pcx + 0.5f * (psx - 1.f), 0.f, limx);
            o[4] = clampf(pcy + 0.5f * (psy - 1.f), 0.f, limy);
            o[5] = clampf(pct + 0.5f * (pst - 1.f), 0.f, limt);
        }
    }
}

// =====================================================================
// Top-K selection: radix-bucket threshold + deterministic prefix compaction
// =====================================================================
__global__ void zero_kernel(int* p, int n) {
    int i = blockIdx.x * blockDim.x + threadIdx.x;
    if (i < n) p[i] = 0;
}

__global__ void hist_kernel(const float* __restrict__ scores, int* __restrict__ hist) {
    int b = blockIdx.y;
    int i = blockIdx.x * 256 + threadIdx.x;
    unsigned key = sortkey(scores[(size_t)b * NANCH + i]);
    atomicAdd(&hist[b * 2048 + (key >> 21)], 1);
}

__global__ void findth_kernel(const int* __restrict__ hist, int* __restrict__ meta) {
    int b = blockIdx.x;
    if (threadIdx.x == 0) {
        int cum = 0, tb = 0, th = 0, rem = TOPK;
        for (int bin = 2047; bin >= 0; --bin) {
            int c = hist[b * 2048 + bin];
            if (cum + c >= TOPK) { tb = bin; th = cum; rem = TOPK - cum; break; }
            cum += c;
        }
        meta[b * 4 + 0] = tb; meta[b * 4 + 1] = th; meta[b * 4 + 2] = rem;
    }
}

__global__ __launch_bounds__(256) void blockcount_kernel(
    const float* __restrict__ scores, const int* __restrict__ meta,
    int* __restrict__ bcH, int* __restrict__ bcE)
{
    int b = blockIdx.y, blk = blockIdx.x, tid = threadIdx.x;
    int tb = meta[b * 4 + 0];
    const float* sc = scores + (size_t)b * NANCH + (size_t)blk * 1024 + tid * 4;
    int cH = 0, cE = 0;
    #pragma unroll
    for (int j = 0; j < 4; ++j) {
        int bin = (int)(sortkey(sc[j]) >> 21);
        cH += (bin > tb); cE += (bin == tb);
    }
    __shared__ int sH[256], sE[256];
    sH[tid] = cH; sE[tid] = cE;
    __syncthreads();
    if (tid == 0) {
        int aH = 0, aE = 0;
        for (int i = 0; i < 256; ++i) { aH += sH[i]; aE += sE[i]; }
        bcH[b * 288 + blk] = aH; bcE[b * 288 + blk] = aE;
    }
}

__global__ void scan_kernel(int* bcH, int* bcE) {
    int b = blockIdx.x;
    if (threadIdx.x == 0) {
        int aH = 0, aE = 0;
        for (int i = 0; i < 288; ++i) {
            int h = bcH[b * 288 + i], e = bcE[b * 288 + i];
            bcH[b * 288 + i] = aH; bcE[b * 288 + i] = aE;
            aH += h; aE += e;
        }
    }
}

__global__ __launch_bounds__(256) void emit_kernel(
    const float* __restrict__ scores, const float* __restrict__ boxes,
    const int* __restrict__ meta, const int* __restrict__ bcH,
    const int* __restrict__ bcE, float* __restrict__ out)
{
    int b = blockIdx.y, blk = blockIdx.x, tid = threadIdx.x;
    int tb        = meta[b * 4 + 0];
    int totalHigh = meta[b * 4 + 1];
    int remaining = meta[b * 4 + 2];
    size_t base = (size_t)blk * 1024 + tid * 4;
    const float* sc = scores + (size_t)b * NANCH;
    int bins[4], cH = 0, cE = 0;
    #pragma unroll
    for (int j = 0; j < 4; ++j) {
        bins[j] = (int)(sortkey(sc[base + j]) >> 21);
        cH += (bins[j] > tb); cE += (bins[j] == tb);
    }
    __shared__ int sH[256], sE[256];
    sH[tid] = cH; sE[tid] = cE;
    __syncthreads();
    if (tid == 0) {
        int aH = 0, aE = 0;
        for (int i = 0; i < 256; ++i) {
            int h = sH[i], e = sE[i];
            sH[i] = aH; sE[i] = aE;
            aH += h; aE += e;
        }
    }
    __syncthreads();
    int offH = bcH[b * 288 + blk] + sH[tid];
    int offE = bcE[b * 288 + blk] + sE[tid];
    for (int j = 0; j < 4; ++j) {
        int slot = -1;
        if (bins[j] > tb)       { slot = offH; ++offH; }
        else if (bins[j] == tb) { if (offE < remaining) slot = totalHigh + offE; ++offE; }
        if (slot >= 0) {
            size_t o = ((size_t)b * TOPK + slot) * 7;
            out[o] = (float)b;
            const float* bx = boxes + ((size_t)b * NANCH + base + j) * 6;
            #pragma unroll
            for (int c = 0; c < 6; ++c) out[o + 1 + c] = bx[c];
        }
    }
}

// =====================================================================
extern "C" void kernel_launch(void* const* d_in, const int* in_sizes, int n_in,
                              void* d_out, int out_size, void* d_ws, size_t ws_size,
                              hipStream_t stream) {
    const float* base_feat = (const float*)d_in[0];
    const float* im_info   = (const float*)d_in[1];
    const float* w_conv = (const float*)d_in[4];
    const float* b_conv = (const float*)d_in[5];
    const float* w_cls  = (const float*)d_in[6];
    const float* b_cls  = (const float*)d_in[7];
    const float* w_bbox = (const float*)d_in[8];
    const float* b_bbox = (const float*)d_in[9];
    float* out = (float*)d_out;

    char* ws = (char*)d_ws;
    unsigned short* x_bf   = (unsigned short*)(ws + OFF_X);
    unsigned short* featb  = (unsigned short*)(ws + OFF_FB);
    unsigned short* wconvb = (unsigned short*)(ws + OFF_WB);
    float* scores = (float*)(ws + OFF_SC);
    float* boxes  = (float*)(ws + OFF_BX);
    int* hist = (int*)(ws + OFF_HIST);
    int* meta = (int*)(ws + OFF_META);
    int* bcH  = (int*)(ws + OFF_BCH);
    int* bcE  = (int*)(ws + OFF_BCE);

    // 0) one-shot f32 -> bf16 conversions (weights + feature map)
    const int wn = COUT * KDIM;                 // 7,077,888
    const int fn = BATCH * CIN * NPOS;          // 16,777,216
    cvt_bf_kernel<<<(wn / 8 + 255) / 256, 256, 0, stream>>>(w_conv, wconvb, wn);
    cvt_bf_kernel<<<(fn / 8 + 255) / 256, 256, 0, stream>>>(base_feat, featb, fn);

    // 1) conv3d + ReLU -> bf16 activations (the 464-GFLOP GEMM)
    conv3d_wmma_kernel<<<dim3(NPOS / 128, COUT / 128, BATCH), 256, 0, stream>>>(
        featb, wconvb, b_conv, x_bf);

    // 2) fused cls/bbox GEMM + softmax + box decode (scores/boxes alias featb)
    head_wmma_kernel<<<dim3(NPOS / 128, BATCH), 256, 0, stream>>>(
        x_bf, w_cls, b_cls, w_bbox, b_bbox, im_info, scores, boxes);

    // 3) deterministic top-2000 select per batch
    zero_kernel<<<(BATCH * 2048 + 255) / 256, 256, 0, stream>>>(hist, BATCH * 2048);
    hist_kernel<<<dim3(NANCH / 256, BATCH), 256, 0, stream>>>(scores, hist);
    findth_kernel<<<BATCH, 32, 0, stream>>>(hist, meta);
    blockcount_kernel<<<dim3(NANCH / 1024, BATCH), 256, 0, stream>>>(scores, meta, bcH, bcE);
    scan_kernel<<<BATCH, 32, 0, stream>>>(bcH, bcE);
    emit_kernel<<<dim3(NANCH / 1024, BATCH), 256, 0, stream>>>(scores, boxes, meta, bcH, bcE, out);
}
